// MiniGPT_73443940761856
// MI455X (gfx1250) — matrix operand order repaired and checked by reading, not compiled, not run on hardware
//
#include <hip/hip_runtime.h>
#include <hip/hip_fp16.h>
#include <math.h>

// ---------------- types for WMMA fragments ----------------
typedef __attribute__((ext_vector_type(16))) _Float16 v16h;
typedef __attribute__((ext_vector_type(8)))  float    v8f;
typedef __attribute__((ext_vector_type(4)))  int      i4;

union Frag { v16h h; i4 i[2]; };

// ---------------- model dims ----------------
#define NV   50257
#define NT   2048
#define NB   2
#define NL   6
#define NH   8
#define NC   512
#define ND   64
#define NFF  2048
#define NROW (NB*NT)   // 4096

// padded LDS strides (avoid 0-mod-64-bank row strides)
#define S_STRIDE 2064      // floats per S row
#define P_STRIDE 2056      // halves per P row
#define V_STRIDE 40        // halves per Vt row
#define B_STRIDE 40        // halves per Bs row

// ---------------- f32 -> f16 convert ----------------
__global__ __launch_bounds__(256) void cvt_f16(const float* __restrict__ s,
                                               _Float16* __restrict__ d, long n) {
  long i = (long)blockIdx.x * blockDim.x + threadIdx.x;
  long stride = (long)gridDim.x * blockDim.x;
  for (; i < n; i += stride) d[i] = (_Float16)s[i];
}

// ---------------- embedding: x = wte[idx] + wpe[pos] ----------------
__global__ __launch_bounds__(128) void embed_kernel(const int* __restrict__ idx,
                                                    const float* __restrict__ wte,
                                                    const float* __restrict__ wpe,
                                                    float* __restrict__ X) {
  int row = blockIdx.x;              // 0..4095
  int t = row & (NT - 1);
  int tok = idx[row];
  const float* we = wte + (size_t)tok * NC;
  const float* pe = wpe + (size_t)t * NC;
  float* xr = X + (size_t)row * NC;
#pragma unroll
  for (int i = 0; i < NC / 128; ++i) {
    int e = threadIdx.x + i * 128;
    xr[e] = we[e] + pe[e];
  }
}

// ---------------- layernorm (f32 in, f16 out) ----------------
__global__ __launch_bounds__(128) void layernorm_f16(const float* __restrict__ X,
                                                     const float* __restrict__ W,
                                                     const float* __restrict__ Bi,
                                                     _Float16* __restrict__ Out) {
  __shared__ float s_sum[4], s_sq[4];
  int row = blockIdx.x;
  const float* xr = X + (size_t)row * NC;
  float v[NC / 128];
  float sum = 0.f, sq = 0.f;
#pragma unroll
  for (int i = 0; i < NC / 128; ++i) {
    v[i] = xr[threadIdx.x + i * 128];
    sum += v[i];
    sq += v[i] * v[i];
  }
#pragma unroll
  for (int off = 16; off > 0; off >>= 1) {
    sum += __shfl_xor(sum, off, 32);
    sq  += __shfl_xor(sq,  off, 32);
  }
  int wave = threadIdx.x >> 5;
  if ((threadIdx.x & 31) == 0) { s_sum[wave] = sum; s_sq[wave] = sq; }
  __syncthreads();
  sum = s_sum[0] + s_sum[1] + s_sum[2] + s_sum[3];
  sq  = s_sq[0]  + s_sq[1]  + s_sq[2]  + s_sq[3];
  float mu = sum * (1.0f / NC);
  float var = sq * (1.0f / NC) - mu * mu;
  float rstd = rsqrtf(var + 1e-5f);
  _Float16* orow = Out + (size_t)row * NC;
#pragma unroll
  for (int i = 0; i < NC / 128; ++i) {
    int e = threadIdx.x + i * 128;
    orow[e] = (_Float16)((v[i] - mu) * rstd * W[e] + Bi[e]);
  }
}

// ---------------- generic WMMA GEMM: out[m][n] = sum_k A[m][k] * B[n][k] ----
// A: M x K f16 row-major, B: N x K f16 row-major.
// Workgroup = 8 waves; WG tile 128(M) x 64(N); each wave one 16x64 strip.
// B chunk (64 x 32) staged per WG into double-buffered LDS via
// CDNA5 async global->LDS copies (ASYNCcnt), overlapped with WMMA compute.
enum { EPI_F16 = 0, EPI_F32 = 1, EPI_RES = 2, EPI_GELU = 3 };

template <int EPI>
__global__ __launch_bounds__(256) void gemm_wmma(const _Float16* __restrict__ A,
                                                 const _Float16* __restrict__ B,
                                                 void* Out, const float* Res,
                                                 int M, int N, int K) {
  __shared__ _Float16 Bs[2][64 * B_STRIDE];   // 2 x 5120 B

  const int tid  = threadIdx.x;
  const int lane = tid & 31;
  const int wave = tid >> 5;
  const int l16  = lane & 15;
  const int sel  = lane >> 4;          // 0 for lanes 0-15, 1 for lanes 16-31
  const int m0 = blockIdx.y * 128 + wave * 16;
  const int n0 = blockIdx.x * 64;

  // staging assignment: thread -> (row n, 8-half k part); one b128 per thread
  const int sn = tid >> 2;             // 0..63
  const int sk = (tid & 3) * 8;        // 0,8,16,24
  int snc = n0 + sn; if (snc >= N) snc = N - 1;   // clamp ragged N (lm_head)
  const _Float16* sbp = B + (size_t)snc * K + sk;

  const unsigned lds0 = (unsigned)(unsigned long long)&Bs[0][sn * B_STRIDE + sk];
  const unsigned lds1 = (unsigned)(unsigned long long)&Bs[1][sn * B_STRIDE + sk];

  v8f acc[4] = {};
  const _Float16* arow = A + (size_t)(m0 + l16) * K;

  const int nch = K / 32;
  // prologue: async-stage chunk 0 into buffer 0
  asm volatile("global_load_async_to_lds_b128 %0, %1, off"
               :: "v"(lds0), "v"(sbp) : "memory");

  for (int c = 0; c < nch; ++c) {
    asm volatile("s_wait_asynccnt 0x0" ::: "memory");  // this wave's copy done
    __syncthreads();                                   // all waves' copies visible
    if (c + 1 < nch) {
      unsigned dst = ((c + 1) & 1) ? lds1 : lds0;
      const _Float16* src = sbp + (size_t)(c + 1) * 32;
      asm volatile("global_load_async_to_lds_b128 %0, %1, off"
                   :: "v"(dst), "v"(src) : "memory");
    }
    const int k = c * 32;
    Frag a;
    const _Float16* ap = arow + k + (sel ? 8 : 0);
    a.i[0] = *(const i4*)(ap);
    a.i[1] = *(const i4*)(ap + 16);
    const _Float16* bs = &Bs[c & 1][0];
#pragma unroll
    for (int j = 0; j < 4; ++j) {
      const _Float16* bp = bs + (j * 16 + l16) * B_STRIDE + (sel ? 8 : 0);
      Frag b;
      b.i[0] = *(const i4*)(bp);
      b.i[1] = *(const i4*)(bp + 16);
      acc[j] = __builtin_amdgcn_wmma_f32_16x16x32_f16(false, a.h, false, b.h,
                                                      (short)0, acc[j], false, false);
    }
  }

#pragma unroll
  for (int j = 0; j < 4; ++j) {
    int n = n0 + j * 16 + l16;
    if (n >= N) continue;
#pragma unroll
    for (int r = 0; r < 8; ++r) {
      int m = m0 + r + sel * 8;
      float v = acc[j][r];
      size_t off = (size_t)m * N + n;
      if (EPI == EPI_F32) {
        ((float*)Out)[off] = v;
      } else if (EPI == EPI_F16) {
        ((_Float16*)Out)[off] = (_Float16)v;
      } else if (EPI == EPI_RES) {
        ((float*)Out)[off] = v + Res[off];
      } else { // EPI_GELU (exact, erf-based), f16 out
        float g = 0.5f * v * (1.0f + erff(v * 0.70710678118654752f));
        ((_Float16*)Out)[off] = (_Float16)g;
      }
    }
  }
}

// ---------------- attention: per-WG (b, h, 16-row tile) -------------------
// qkv: [4096][1536] f16 (q|k|v each 512). y: [4096][512] f16.
// 4 waves; S 16 x S_STRIDE f32 LDS, P 16 x P_STRIDE f16 LDS,
// Vt 2 x 64 x V_STRIDE f16 LDS (double-buffered transposed V staging).
__global__ __launch_bounds__(128) void attention_kernel(const _Float16* __restrict__ qkv,
                                                        _Float16* __restrict__ y) {
  extern __shared__ char smem_raw[];
  float*    S   = (float*)smem_raw;                                  // 16 x 2064 f32
  _Float16* P   = (_Float16*)(smem_raw + 16 * S_STRIDE * 4);         // 16 x 2056 f16
  _Float16* VtB = (_Float16*)(smem_raw + 16 * S_STRIDE * 4 + 16 * P_STRIDE * 2);
  __shared__ float red[16][8];

  const int tid  = threadIdx.x;
  const int lane = tid & 31;
  const int wave = tid >> 5;             // 0..3
  const int l16 = lane & 15;
  const int sel = lane >> 4;
  const int tile = blockIdx.x;           // 0..127
  const int h = blockIdx.y;
  const int b = blockIdx.z;
  const int t0 = tile * 16;
  const size_t rowbase = (size_t)b * NT;
  const int ldq = 3 * NC;                // 1536

  // ---- Q fragments: 16 x 64, two K-chunks of 32 ----
  Frag qa[2];
  {
    const _Float16* qrow = qkv + (rowbase + t0 + l16) * ldq + h * ND;
#pragma unroll
    for (int c = 0; c < 2; ++c) {
      const _Float16* p = qrow + c * 32 + (sel ? 8 : 0);
      qa[c].i[0] = *(const i4*)(p);
      qa[c].i[1] = *(const i4*)(p + 16);
    }
  }

  // ---- scores S = (Q K^T) * scale with causal mask ----
  const float scale = 0.125f;            // 1/sqrt(64)
  for (int st = wave; st < NT / 16; st += 4) {
    v8f acc = {};
#pragma unroll
    for (int c = 0; c < 2; ++c) {
      const _Float16* kp = qkv + (rowbase + st * 16 + l16) * ldq + NC + h * ND
                           + c * 32 + (sel ? 8 : 0);
      Frag kb;
      kb.i[0] = *(const i4*)(kp);
      kb.i[1] = *(const i4*)(kp + 16);
      acc = __builtin_amdgcn_wmma_f32_16x16x32_f16(false, qa[c].h, false, kb.h,
                                                   (short)0, acc, false, false);
    }
#pragma unroll
    for (int r = 0; r < 8; ++r) {
      int m = r + sel * 8;
      int sg = st * 16 + l16;
      float v = acc[r] * scale;
      if (sg > t0 + m) v = -__builtin_inff();
      S[m * S_STRIDE + sg] = v;
    }
  }
  __syncthreads();

  // ---- softmax over each of the 16 rows (8 threads per row, stride-8) ----
  {
    int row = tid >> 3;                  // 0..15
    int seg = tid & 7;                   // 0..7
    float* srow = S + row * S_STRIDE;
    float mx = -__builtin_inff();
    for (int i = 0; i < 256; ++i) mx = fmaxf(mx, srow[i * 8 + seg]);
    red[row][seg] = mx;
    __syncthreads();
    mx = red[row][0];
#pragma unroll
    for (int i = 1; i < 8; ++i) mx = fmaxf(mx, red[row][i]);
    __syncthreads();                     // before reusing red for sums
    float sum = 0.f;
    for (int i = 0; i < 256; ++i) {
      int jx = i * 8 + seg;
      float e = __expf(srow[jx] - mx);
      srow[jx] = e;
      sum += e;
    }
    red[row][seg] = sum;
    __syncthreads();
    sum = 0.f;
#pragma unroll
    for (int i = 0; i < 8; ++i) sum += red[row][i];
    float inv = 1.0f / sum;
    _Float16* prow = P + row * P_STRIDE;
    for (int i = 0; i < 256; ++i) {
      int jx = i * 8 + seg;
      prow[jx] = (_Float16)(srow[jx] * inv);
    }
  }
  __syncthreads();

  // ---- O = P V : double-buffered transposed V staging in LDS -------------
  const int dbase = wave * 16;
  const int vr    = tid >> 2;            // 0..31 : s row within chunk
  const int vpart = (tid & 3) * 16;      // d group of 16
  const _Float16* vbase = qkv + rowbase * ldq + 2 * NC + h * ND + vpart;

  // prologue: stage chunk 0 into buffer 0
  {
    Frag tmp;
    const _Float16* vsrc = vbase + (size_t)vr * ldq;
    tmp.i[0] = *(const i4*)(vsrc);
    tmp.i[1] = *(const i4*)(vsrc + 8);
    _Float16* vt = VtB;
#pragma unroll
    for (int e = 0; e < 16; ++e) vt[(vpart + e) * V_STRIDE + vr] = tmp.h[e];
  }

  v8f oacc = {};
  for (int c = 0; c < NT / 32; ++c) {
    __syncthreads();                     // Vt[c&1] ready; prior reads finished
    Frag nxt;
    const bool more = (c + 1 < NT / 32);
    if (more) {
      const _Float16* vsrc = vbase + (size_t)((c + 1) * 32 + vr) * ldq;
      nxt.i[0] = *(const i4*)(vsrc);
      nxt.i[1] = *(const i4*)(vsrc + 8);
    }
    const _Float16* vt = VtB + (c & 1) * (64 * V_STRIDE);
    Frag pa;
    const _Float16* pp = P + l16 * P_STRIDE + c * 32 + (sel ? 8 : 0);
    pa.i[0] = *(const i4*)(pp);
    pa.i[1] = *(const i4*)(pp + 16);
    Frag vb;
    const _Float16* vp = vt + (dbase + l16) * V_STRIDE + (sel ? 8 : 0);
    vb.i[0] = *(const i4*)(vp);
    vb.i[1] = *(const i4*)(vp + 16);
    oacc = __builtin_amdgcn_wmma_f32_16x16x32_f16(false, pa.h, false, vb.h,
                                                  (short)0, oacc, false, false);
    if (more) {
      _Float16* vtn = VtB + ((c + 1) & 1) * (64 * V_STRIDE);
#pragma unroll
      for (int e = 0; e < 16; ++e) vtn[(vpart + e) * V_STRIDE + vr] = nxt.h[e];
    }
  }
#pragma unroll
  for (int r = 0; r < 8; ++r) {
    int m = r + sel * 8;
    y[(rowbase + t0 + m) * NC + h * ND + dbase + l16] = (_Float16)oacc[r];
  }
}

// ---------------- workspace layout (bytes) ----------------
#define OFF_WTE16   ((size_t)0)
#define SZ_WTE16    ((size_t)NV * NC * 2)
#define OFF_ATTN16  (OFF_WTE16 + SZ_WTE16)
#define SZ_ATTN16   ((size_t)NL * 3 * NC * NC * 2)
#define OFF_CPROJ16 (OFF_ATTN16 + SZ_ATTN16)
#define SZ_CPROJ16  ((size_t)NL * NC * NC * 2)
#define OFF_FC16    (OFF_CPROJ16 + SZ_CPROJ16)
#define SZ_FC16     ((size_t)NL * NFF * NC * 2)
#define OFF_PROJ16  (OFF_FC16 + SZ_FC16)
#define SZ_PROJ16   ((size_t)NL * NC * NFF * 2)
#define OFF_X       (OFF_PROJ16 + SZ_PROJ16)
#define SZ_X        ((size_t)NROW * NC * 4)
#define OFF_XN      (OFF_X + SZ_X)
#define SZ_XN       ((size_t)NROW * NC * 2)
#define OFF_QKV     (OFF_XN + SZ_XN)
#define SZ_QKV      ((size_t)NROW * 3 * NC * 2)
#define OFF_Y       (OFF_QKV + SZ_QKV)
#define SZ_Y        ((size_t)NROW * NC * 2)
#define OFF_H       (OFF_Y + SZ_Y)
#define SZ_H        ((size_t)NROW * NFF * 2)

extern "C" void kernel_launch(void* const* d_in, const int* in_sizes, int n_in,
                              void* d_out, int out_size, void* d_ws, size_t ws_size,
                              hipStream_t stream) {
  (void)in_sizes; (void)n_in; (void)out_size; (void)ws_size;

  const int*   idx    = (const int*)  d_in[0];
  const float* wte    = (const float*)d_in[1];
  const float* wpe    = (const float*)d_in[2];
  const float* ln1w   = (const float*)d_in[3];
  const float* ln1b   = (const float*)d_in[4];
  const float* attnw  = (const float*)d_in[5];
  const float* cprojw = (const float*)d_in[6];
  const float* ln2w   = (const float*)d_in[7];
  const float* ln2b   = (const float*)d_in[8];
  const float* fcw    = (const float*)d_in[9];
  const float* projw  = (const float*)d_in[10];
  const float* lnfw   = (const float*)d_in[11];
  const float* lnfb   = (const float*)d_in[12];
  float* out = (float*)d_out;

  char* ws = (char*)d_ws;
  _Float16* wte16   = (_Float16*)(ws + OFF_WTE16);
  _Float16* attn16  = (_Float16*)(ws + OFF_ATTN16);
  _Float16* cproj16 = (_Float16*)(ws + OFF_CPROJ16);
  _Float16* fc16    = (_Float16*)(ws + OFF_FC16);
  _Float16* proj16  = (_Float16*)(ws + OFF_PROJ16);
  float*    x       = (float*)   (ws + OFF_X);
  _Float16* xn16    = (_Float16*)(ws + OFF_XN);
  _Float16* qkv16   = (_Float16*)(ws + OFF_QKV);
  _Float16* y16     = (_Float16*)(ws + OFF_Y);
  _Float16* h16     = (_Float16*)(ws + OFF_H);

  // 1) convert all weights to f16
  cvt_f16<<<2048, 256, 0, stream>>>(wte,    wte16,   (long)NV * NC);
  cvt_f16<<<2048, 256, 0, stream>>>(attnw,  attn16,  (long)NL * 3 * NC * NC);
  cvt_f16<<<2048, 256, 0, stream>>>(cprojw, cproj16, (long)NL * NC * NC);
  cvt_f16<<<2048, 256, 0, stream>>>(fcw,    fc16,    (long)NL * NFF * NC);
  cvt_f16<<<2048, 256, 0, stream>>>(projw,  proj16,  (long)NL * NC * NFF);

  // 2) embedding
  embed_kernel<<<NROW, 128, 0, stream>>>(idx, wte, wpe, x);

  const dim3 gemm_blk(256);
  const int  MB = NROW / 128;                       // 32 M-blocks
  const size_t att_smem = (size_t)16 * S_STRIDE * 4 + (size_t)16 * P_STRIDE * 2
                        + (size_t)2 * 64 * V_STRIDE * 2;

  // 3) transformer layers
  for (int l = 0; l < NL; ++l) {
    const _Float16* wa  = attn16  + (size_t)l * 3 * NC * NC;
    const _Float16* wcp = cproj16 + (size_t)l * NC * NC;
    const _Float16* wfc = fc16    + (size_t)l * NFF * NC;
    const _Float16* wpj = proj16  + (size_t)l * NC * NFF;

    layernorm_f16<<<NROW, 128, 0, stream>>>(x, ln1w + l * NC, ln1b + l * NC, xn16);
    gemm_wmma<EPI_F16><<<dim3(3 * NC / 64, MB), gemm_blk, 0, stream>>>(
        xn16, wa, qkv16, nullptr, NROW, 3 * NC, NC);
    attention_kernel<<<dim3(NT / 16, NH, NB), 128, att_smem, stream>>>(qkv16, y16);
    gemm_wmma<EPI_RES><<<dim3(NC / 64, MB), gemm_blk, 0, stream>>>(
        y16, wcp, x, x, NROW, NC, NC);

    layernorm_f16<<<NROW, 128, 0, stream>>>(x, ln2w + l * NC, ln2b + l * NC, xn16);
    gemm_wmma<EPI_GELU><<<dim3(NFF / 64, MB), gemm_blk, 0, stream>>>(
        xn16, wfc, h16, nullptr, NROW, NFF, NC);
    gemm_wmma<EPI_RES><<<dim3(NC / 64, MB), gemm_blk, 0, stream>>>(
        h16, wpj, x, x, NROW, NC, NFF);
  }

  // 4) final LN + tied lm_head
  layernorm_f16<<<NROW, 128, 0, stream>>>(x, lnfw, lnfb, xn16);
  gemm_wmma<EPI_F32><<<dim3((NV + 63) / 64, MB), gemm_blk, 0, stream>>>(
      xn16, wte16, out, nullptr, NROW, NV, NC);
}
